// GVPModel_72980084294215
// MI455X (gfx1250) — matrix operand order repaired and checked by reading, not compiled, not run on hardware
//
#include <hip/hip_runtime.h>
#include <hip/hip_bf16.h>

// ---------------------------------------------------------------------------
// GVP-GNN forward for MI455X (gfx1250, wave32).
// Heavy per-edge / per-node GVP matmuls run as bf16 WMMA GEMMs
// (v_wmma_f32_16x16x32_bf16, f32 accumulation). Everything streams through a
// bump-allocated workspace in d_ws.
//
// Param layout assumption (compile-only; cannot verify): the harness flattens
// the params pytree in JAX order (dicts by sorted key, lists by index), giving
// 73 float32 leaves at d_in[5..77]:
//   W_e{wh,ws_b,ws_w,wsv_b,wsv_w,wv}, W_out{wh,ws_b,ws_w}, W_v{ws_b,ws_w},
//   layers[0..1]{ff0{6},ff1{6},ln0{b,w},ln1{b,w},msg0{6},msg1{6}},
//   ln_e{b,w}, ln_out{b,w}, ln_v{b,w}
// ---------------------------------------------------------------------------

typedef __bf16 bf16_t;
typedef __attribute__((ext_vector_type(16))) __bf16 v16bf;
typedef __attribute__((ext_vector_type(8)))  float  v8f;

static __device__ __forceinline__ bf16_t f2bf(float x) { return (bf16_t)x; }

// ---------------------------------------------------------------------------
// Generic bf16 GEMM: C[M,Np] = A[M,Kp] * B[Kp,Np], f32 accum via WMMA.
// One wave per 16x16 C tile. A row-major (lda), B row-major (ldb=Np), padded
// to Kp % 32 == 0, Np % 16 == 0, M % 16 == 0.
// Fragment layouts used here (A and B share one contiguous k order):
//   A 16x32 bf16 : lanes 0-15 row M=lane, k 0..15; lanes 16-31 same
//                  row, k 16..31; VGPR i holds pair (k, k+1).
//   B 32x16 bf16 : lanes 0-15 col N=lane, k 0..15; lanes 16-31 k 16..31.
//   C 16x16 f32  : VGPR r -> row r (lanes 0-15) / row 8+r (lanes 16-31).
// ---------------------------------------------------------------------------
__global__ void gemm_bf16_wave(const bf16_t* __restrict__ A, int lda,
                               const bf16_t* __restrict__ B, int ldb,
                               float* __restrict__ C, int ldc,
                               int Mtiles, int Ntiles, int Ksteps) {
  int gw   = (int)((blockIdx.x * blockDim.x + threadIdx.x) >> 5);
  int lane = (int)(threadIdx.x & 31);
  if (gw >= Mtiles * Ntiles) return;          // wave-uniform exit (EXEC all-1 for WMMA)
  int nt = gw % Ntiles;
  int mt = gw / Ntiles;
  int half = lane >> 4;                        // 0 or 1
  int l15  = lane & 15;
  int arow = (mt << 4) + l15;
  int bcol = (nt << 4) + l15;
  int kA = half << 4;                          // A k-offset: 0 or 16
  int kB = half << 4;                          // B k-offset: 0 or 16
  v8f acc = {};
  const bf16_t* Ap = A + (size_t)arow * lda + kA;
  const bf16_t* Bp = B + (size_t)kB * ldb + bcol;
  for (int ks = 0; ks < Ksteps; ++ks) {
    v16bf a{}, b{};
#pragma unroll
    for (int i = 0; i < 8; ++i) {
      int kk = i << 1;                             // 0,2,4,...,14
      a[2 * i]     = Ap[kk];
      a[2 * i + 1] = Ap[kk + 1];
    }
#pragma unroll
    for (int j = 0; j < 8; ++j) {
      b[2 * j]     = Bp[(size_t)(2 * j) * ldb];
      b[2 * j + 1] = Bp[(size_t)(2 * j + 1) * ldb];
    }
    if (ks + 1 < Ksteps) __builtin_prefetch(Ap + 32, 0, 1);  // global_prefetch_b8
    acc = __builtin_amdgcn_wmma_f32_16x16x32_bf16(
        /*neg_a=*/false, a, /*neg_b=*/false, b,
        /*c_mod=*/(short)0, acc, /*reuse_a=*/false, /*reuse_b=*/false);
    Ap += 32;
    Bp += (size_t)32 * ldb;
  }
  int crow = (mt << 4) + (half << 3);
  float* Cp = C + (size_t)crow * ldc + bcol;
#pragma unroll
  for (int r = 0; r < 8; ++r) Cp[(size_t)r * ldc] = acc[r];
}

// Cast an f32 weight (K0 x N0) into zero-padded bf16 (Kp x Np).
__global__ void k_pack_weight(const float* __restrict__ W, int K0, int N0,
                              bf16_t* __restrict__ Wp, int Kp, int Np) {
  int i = blockIdx.x * blockDim.x + threadIdx.x;
  if (i >= Kp * Np) return;
  int k = i / Np, n = i % Np;
  float v = (k < K0 && n < N0) ? W[k * N0 + n] : 0.0f;
  Wp[i] = f2bf(v);
}

// Node embedding: LayerNorm(attrs,5) then W_v (5->128), hv starts at zero.
__global__ void k_node_init(const float* __restrict__ attrs,
                            const float* __restrict__ lnw, const float* __restrict__ lnb,
                            const float* __restrict__ Ww, const float* __restrict__ Wb,
                            float* __restrict__ hs, bf16_t* __restrict__ hs_bf, int NN) {
  int n = blockIdx.x * blockDim.x + threadIdx.x;
  if (n >= NN) return;
  float a[5];
  float mu = 0.f;
  for (int i = 0; i < 5; ++i) { a[i] = attrs[n * 5 + i]; mu += a[i]; }
  mu *= 0.2f;
  float var = 0.f;
  for (int i = 0; i < 5; ++i) { float d = a[i] - mu; var += d * d; }
  var *= 0.2f;
  float inv = rsqrtf(var + 1e-5f);
  for (int i = 0; i < 5; ++i) a[i] = (a[i] - mu) * inv * lnw[i] + lnb[i];
  for (int j = 0; j < 128; ++j) {
    float s = Wb[j];
    for (int i = 0; i < 5; ++i) s += a[i] * Ww[i * 128 + j];
    hs[(size_t)n * 128 + j] = s;
    hs_bf[(size_t)n * 128 + j] = f2bf(s);
  }
}

// Edge features: geometry, Bessel radial basis + poly envelope, LN, W_e GVP.
__global__ void k_edge_init(const float* __restrict__ pos,
                            const int* __restrict__ src, const int* __restrict__ dst,
                            const float* __restrict__ shifts,
                            const float* __restrict__ lnw, const float* __restrict__ lnb,
                            const float* __restrict__ wh, const float* __restrict__ wsw,
                            const float* __restrict__ wsb, const float* __restrict__ wsvw,
                            const float* __restrict__ wsvb, const float* __restrict__ wv,
                            bf16_t* __restrict__ es_bf, bf16_t* __restrict__ ev_bf,
                            size_t ps_ev, int E) {
  int e = blockIdx.x * blockDim.x + threadIdx.x;
  if (e >= E) return;
  int s = src[e], d = dst[e];
  float vx = pos[3 * d + 0] - pos[3 * s + 0] + shifts[3 * e + 0];
  float vy = pos[3 * d + 1] - pos[3 * s + 1] + shifts[3 * e + 1];
  float vz = pos[3 * d + 2] - pos[3 * s + 2] + shifts[3 * e + 2];
  float r  = sqrtf(vx * vx + vy * vy + vz * vz);
  float ir = 1.0f / (r + 1e-9f);
  float ux = vx * ir, uy = vy * ir, uz = vz * ir;
  float x  = r * 0.2f;                         // r / CUTOFF
  float x2 = x * x, x6 = x2 * x2 * x2;
  float env = 1.0f - 28.0f * x6 + 48.0f * x6 * x - 21.0f * x6 * x2;  // p=6 poly
  if (x >= 1.0f) env = 0.0f;
  float es[8];
  float c0 = sqrtf(0.4f) * env / r;
  for (int i = 0; i < 8; ++i)
    es[i] = c0 * __sinf((float)(i + 1) * 3.14159265358979f * x);
  float mu = 0.f;
  for (int i = 0; i < 8; ++i) mu += es[i];
  mu *= 0.125f;
  float var = 0.f;
  for (int i = 0; i < 8; ++i) { float t = es[i] - mu; var += t * t; }
  var *= 0.125f;
  float inv = rsqrtf(var + 1e-5f);
  for (int i = 0; i < 8; ++i) es[i] = (es[i] - mu) * inv * lnw[i] + lnb[i];
  float n2 = fmaxf(ux * ux + uy * uy + uz * uz, 1e-8f);
  float invn = rsqrtf(n2);                     // vector LayerNorm (single vector)
  ux *= invn; uy *= invn; uz *= invn;
  float w00 = wh[0];
  float hx = ux * w00, hy = uy * w00, hz = uz * w00;
  float vn = sqrtf(fmaxf(hx * hx + hy * hy + hz * hz, 1e-8f));
  float so[32];
  for (int j = 0; j < 32; ++j) {
    float a = wsb[j];
    for (int i = 0; i < 8; ++i) a += es[i] * wsw[i * 32 + j];
    a += vn * wsw[8 * 32 + j];
    so[j] = a;
  }
  float g = wsvb[0];
  for (int j = 0; j < 32; ++j) g += so[j] * wsvw[j];
  g = 1.0f / (1.0f + __expf(-g));
  float wvv = wv[0] * g;
  ev_bf[0 * ps_ev + e] = f2bf(hx * wvv);
  ev_bf[1 * ps_ev + e] = f2bf(hy * wvv);
  ev_bf[2 * ps_ev + e] = f2bf(hz * wvv);
  for (int j = 0; j < 32; ++j) es_bf[(size_t)e * 32 + j] = f2bf(so[j]);
}

// A_s = [hs[src](128) | es(32) | hs[dst](128) | zeros(96)] (E x 384)
__global__ void k_gather_s(const bf16_t* __restrict__ hs_bf, const bf16_t* __restrict__ es_bf,
                           const int* __restrict__ src, const int* __restrict__ dst,
                           bf16_t* __restrict__ A, int E) {
  int i = blockIdx.x * blockDim.x + threadIdx.x;
  if (i >= E * 384) return;
  int e = i / 384, j = i % 384;
  bf16_t v = (bf16_t)0.0f;
  if (j < 128)      v = hs_bf[(size_t)src[e] * 128 + j];
  else if (j < 160) v = es_bf[(size_t)e * 32 + (j - 128)];
  else if (j < 288) v = hs_bf[(size_t)dst[e] * 128 + (j - 160)];
  A[i] = v;
}

// A_v[c] = [hv[src](32) | ev(1) | hv[dst](32) | zeros(31)] (3 planes, E x 96)
__global__ void k_gather_v(const bf16_t* __restrict__ hv_bf, size_t ps_hv,
                           const bf16_t* __restrict__ ev_bf, size_t ps_ev,
                           const int* __restrict__ src, const int* __restrict__ dst,
                           bf16_t* __restrict__ A, size_t ps_av, int E) {
  int i = blockIdx.x * blockDim.x + threadIdx.x;
  if (i >= E * 288) return;
  int e = i / 288, r = i % 288, c = r / 96, j = r % 96;
  bf16_t v = (bf16_t)0.0f;
  if (j < 32)       v = hv_bf[c * ps_hv + (size_t)src[e] * 32 + j];
  else if (j == 32) v = ev_bf[c * ps_ev + e];
  else if (j < 65)  v = hv_bf[c * ps_hv + (size_t)dst[e] * 32 + (j - 33)];
  A[c * ps_av + (size_t)e * 96 + j] = v;
}

// vn = sqrt(max(sum_c vh^2, eps)) written as bf16 into columns of dst.
__global__ void k_vn(const float* __restrict__ C, size_t ps, int ldc, int H,
                     bf16_t* __restrict__ D, int lda, int coff, int M) {
  int i = blockIdx.x * blockDim.x + threadIdx.x;
  if (i >= M * H) return;
  int m = i / H, h = i % H;
  size_t o = (size_t)m * ldc + h;
  float x = C[o], y = C[ps + o], z = C[2 * ps + o];
  D[(size_t)m * lda + coff + h] = f2bf(sqrtf(fmaxf(x * x + y * y + z * z, 1e-8f)));
}

// Cast 3 f32 planes (ldc, H valid cols) to bf16 planes (ldb, zero-padded).
__global__ void k_cast_planes(const float* __restrict__ C, size_t psc, int ldc, int H,
                              bf16_t* __restrict__ B, size_t psb, int ldb, int M) {
  int i = blockIdx.x * blockDim.x + threadIdx.x;
  if (i >= M * 3 * ldb) return;
  int m = i / (3 * ldb), r = i % (3 * ldb), c = r / ldb, j = r % ldb;
  float v = (j < H) ? C[c * psc + (size_t)m * ldc + j] : 0.0f;
  B[c * psb + (size_t)m * ldb + j] = f2bf(v);
}

__global__ void k_copy_bf(const bf16_t* __restrict__ S, int w,
                          bf16_t* __restrict__ D, int lda, int M) {
  int i = blockIdx.x * blockDim.x + threadIdx.x;
  if (i >= M * w) return;
  int m = i / w, j = i % w;
  D[(size_t)m * lda + j] = S[i];
}

// C += bias; act 0=none, 1=silu; optional bf16 mirror.
__global__ void k_bias_act(float* __restrict__ C, const float* __restrict__ b,
                           int M, int N, int act, bf16_t* __restrict__ OB) {
  int i = blockIdx.x * blockDim.x + threadIdx.x;
  if (i >= M * N) return;
  int n = i % N;
  float v = C[i] + b[n];
  if (act == 1) v = v / (1.0f + __expf(-v));
  C[i] = v;
  if (OB) OB[i] = f2bf(v);
}

// V[c] *= sigmoid(G + gb) per element; optional bf16 mirror.
__global__ void k_gate_mul(float* __restrict__ V, size_t psv,
                           const float* __restrict__ G, const float* __restrict__ gb,
                           int M, int N, bf16_t* __restrict__ OB, size_t psb) {
  int i = blockIdx.x * blockDim.x + threadIdx.x;
  if (i >= M * N) return;
  int n = i % N;
  float g = 1.0f / (1.0f + __expf(-(G[i] + gb[n])));
  for (int c = 0; c < 3; ++c) {
    float v = V[c * psv + i] * g;
    V[c * psv + i] = v;
    if (OB) OB[c * psb + i] = f2bf(v);
  }
}

__global__ void k_count(const int* __restrict__ dst, float* __restrict__ cnt, int E) {
  int e = blockIdx.x * blockDim.x + threadIdx.x;
  if (e < E) atomicAdd(&cnt[dst[e]], 1.0f);
}

__global__ void k_scatter_add(const float* __restrict__ X, const int* __restrict__ idx,
                              float* __restrict__ acc, int w, int M) {
  int i = blockIdx.x * blockDim.x + threadIdx.x;
  if (i >= M * w) return;
  int m = i / w, j = i % w;
  atomicAdd(&acc[(size_t)idx[m] * w + j], X[i]);
}

// hs/hv <- LayerNorm(hs + As*ic, hv + Av*ic); optional residual/count/bf16.
__global__ void k_node_ln(float* __restrict__ hs, float* __restrict__ hv, size_t ps_hv,
                          const float* __restrict__ As, const float* __restrict__ Av,
                          size_t ps_av, int ld_av, const float* __restrict__ cnt,
                          const float* __restrict__ lw, const float* __restrict__ lb,
                          bf16_t* __restrict__ hsb, bf16_t* __restrict__ hvb,
                          size_t ps_hvb, int NN) {
  int n = blockIdx.x * blockDim.x + threadIdx.x;
  if (n >= NN) return;
  float ic = 1.0f;
  if (cnt) ic = 1.0f / fmaxf(cnt[n], 1.0f);
  float s[128];
  float mu = 0.f;
  for (int j = 0; j < 128; ++j) {
    float x = hs[(size_t)n * 128 + j];
    if (As) x += As[(size_t)n * 128 + j] * ic;
    s[j] = x; mu += x;
  }
  mu *= (1.0f / 128.0f);
  float var = 0.f;
  for (int j = 0; j < 128; ++j) { float d = s[j] - mu; var += d * d; }
  var *= (1.0f / 128.0f);
  float inv = rsqrtf(var + 1e-5f);
  for (int j = 0; j < 128; ++j) {
    float y = (s[j] - mu) * inv * lw[j] + lb[j];
    hs[(size_t)n * 128 + j] = y;
    if (hsb) hsb[(size_t)n * 128 + j] = f2bf(y);
  }
  float v[3][32];
  float msum = 0.f;
  for (int h = 0; h < 32; ++h) {
    float n2 = 0.f;
    for (int c = 0; c < 3; ++c) {
      float x = hv[c * ps_hv + (size_t)n * 32 + h];
      if (Av) x += Av[c * ps_av + (size_t)n * ld_av + h] * ic;
      v[c][h] = x; n2 += x * x;
    }
    msum += fmaxf(n2, 1e-8f);
  }
  float dn = rsqrtf(msum * (1.0f / 32.0f));
  for (int h = 0; h < 32; ++h)
    for (int c = 0; c < 3; ++c) {
      float y = v[c][h] * dn;
      hv[c * ps_hv + (size_t)n * 32 + h] = y;
      if (hvb) hvb[c * ps_hvb + (size_t)n * 32 + h] = f2bf(y);
    }
}

// Output head: W_out GVP (vo=0) then per-graph atomic sum.
__global__ void k_out_node(const float* __restrict__ hs, const float* __restrict__ hv,
                           size_t ps_hv, const float* __restrict__ wh,
                           const float* __restrict__ wsw, const float* __restrict__ wsb,
                           const int* __restrict__ batch, float* __restrict__ osum,
                           float* __restrict__ ocnt, int NN) {
  int n = blockIdx.x * blockDim.x + threadIdx.x;
  if (n >= NN) return;
  float o0 = wsb[0], o1 = wsb[1], o2 = wsb[2], o3 = wsb[3];
  for (int i = 0; i < 128; ++i) {
    float x = hs[(size_t)n * 128 + i];
    o0 += x * wsw[i * 4 + 0]; o1 += x * wsw[i * 4 + 1];
    o2 += x * wsw[i * 4 + 2]; o3 += x * wsw[i * 4 + 3];
  }
  for (int h = 0; h < 32; ++h) {
    float ax = 0.f, ay = 0.f, az = 0.f;
    for (int v = 0; v < 32; ++v) {
      float w = wh[v * 32 + h];
      ax += hv[0 * ps_hv + (size_t)n * 32 + v] * w;
      ay += hv[1 * ps_hv + (size_t)n * 32 + v] * w;
      az += hv[2 * ps_hv + (size_t)n * 32 + v] * w;
    }
    float vn = sqrtf(fmaxf(ax * ax + ay * ay + az * az, 1e-8f));
    int i = 128 + h;
    o0 += vn * wsw[i * 4 + 0]; o1 += vn * wsw[i * 4 + 1];
    o2 += vn * wsw[i * 4 + 2]; o3 += vn * wsw[i * 4 + 3];
  }
  int g = batch[n];
  atomicAdd(&osum[g * 4 + 0], o0); atomicAdd(&osum[g * 4 + 1], o1);
  atomicAdd(&osum[g * 4 + 2], o2); atomicAdd(&osum[g * 4 + 3], o3);
  atomicAdd(&ocnt[g], 1.0f);
}

__global__ void k_out_div(const float* __restrict__ osum, const float* __restrict__ ocnt,
                          float* __restrict__ out) {
  int i = threadIdx.x;
  if (i < 64) out[i] = osum[i] / fmaxf(ocnt[i >> 2], 1.0f);
}

// ---------------------------------------------------------------------------
extern "C" void kernel_launch(void* const* d_in, const int* in_sizes, int n_in,
                              void* d_out, int out_size, void* d_ws, size_t ws_size,
                              hipStream_t stream) {
  (void)n_in; (void)out_size; (void)ws_size;
  const float* positions  = (const float*)d_in[0];
  const int*   eidx       = (const int*)d_in[1];
  const float* shifts     = (const float*)d_in[2];
  const float* node_attrs = (const float*)d_in[3];
  const int*   batch      = (const int*)d_in[4];
  const float* const* P   = (const float* const*)(d_in + 5);

  const int E  = in_sizes[1] / 2;   // 320000 (multiple of 16)
  const int NN = in_sizes[3] / 5;   // 10000  (multiple of 16)
  const int* src  = eidx;
  const int* dstp = eidx + E;

  char* base = (char*)d_ws;
  size_t off = 0;
  auto alloc  = [&](size_t bytes) -> void* {
    size_t a = (off + 255) & ~(size_t)255;
    off = a + bytes;
    return (void*)(base + a);
  };
  auto allocF = [&](size_t n) { return (float*)alloc(n * 4); };
  auto allocB = [&](size_t n) { return (bf16_t*)alloc(n * 2); };
  auto nblk   = [](long long n) { return (unsigned)((n + 255) / 256); };

  auto pack = [&](const float* W, int K0, int N0, int Kp, int Np) -> bf16_t* {
    bf16_t* Wp = allocB((size_t)Kp * Np);
    k_pack_weight<<<nblk((long long)Kp * Np), 256, 0, stream>>>(W, K0, N0, Wp, Kp, Np);
    return Wp;
  };

  struct LW {
    bf16_t *m0_wh, *m0_ws, *m0_wsv, *m0_wv;
    bf16_t *m1_wh, *m1_ws, *m1_wsv, *m1_wv;
    bf16_t *f0_wh, *f0_ws, *f0_wsv, *f0_wv;
    bf16_t *f1_wh, *f1_ws, *f1_wsv, *f1_wv;
    const float *m0_wsb, *m0_wsvb, *m1_wsb, *m1_wsvb;
    const float *f0_wsb, *f0_wsvb, *f1_wsb, *f1_wsvb;
    const float *ln0_w, *ln0_b, *ln1_w, *ln1_b;
  } L[2];

  for (int l = 0; l < 2; ++l) {
    int b = 11 + 28 * l;
    L[l].f0_wh   = pack(P[b + 0],  32,  64,  32,  64);
    L[l].f0_wsb  = P[b + 1];
    L[l].f0_ws   = pack(P[b + 2], 192, 512, 192, 512);
    L[l].f0_wsvb = P[b + 3];
    L[l].f0_wsv  = pack(P[b + 4], 512,  64, 512,  64);
    L[l].f0_wv   = pack(P[b + 5],  64,  64,  64,  64);
    L[l].f1_wh   = pack(P[b + 6],  64,  64,  64,  64);
    L[l].f1_wsb  = P[b + 7];
    L[l].f1_ws   = pack(P[b + 8], 576, 128, 576, 128);
    L[l].f1_wsvb = P[b + 9];
    L[l].f1_wsv  = pack(P[b + 10], 128, 32, 128,  32);
    L[l].f1_wv   = pack(P[b + 11],  64, 32,  64,  32);
    L[l].ln0_b = P[b + 12]; L[l].ln0_w = P[b + 13];
    L[l].ln1_b = P[b + 14]; L[l].ln1_w = P[b + 15];
    L[l].m0_wh   = pack(P[b + 16],  65,  65,  96,  80);
    L[l].m0_wsb  = P[b + 17];
    L[l].m0_ws   = pack(P[b + 18], 353, 128, 384, 128);
    L[l].m0_wsvb = P[b + 19];
    L[l].m0_wsv  = pack(P[b + 20], 128,  32, 128,  32);
    L[l].m0_wv   = pack(P[b + 21],  65,  32,  96,  32);
    L[l].m1_wh   = pack(P[b + 22],  32,  32,  32,  32);
    L[l].m1_wsb  = P[b + 23];
    L[l].m1_ws   = pack(P[b + 24], 160, 128, 160, 128);
    L[l].m1_wsvb = P[b + 25];
    L[l].m1_wsv  = pack(P[b + 26], 128,  32, 128,  32);
    L[l].m1_wv   = pack(P[b + 27],  32,  32,  32,  32);
  }

  const float* We_wh   = P[0];  const float* We_wsb  = P[1];  const float* We_wsw = P[2];
  const float* We_wsvb = P[3];  const float* We_wsvw = P[4];  const float* We_wv  = P[5];
  const float* Wo_wh   = P[6];  const float* Wo_wsb  = P[7];  const float* Wo_wsw = P[8];
  const float* Wv_wsb  = P[9];  const float* Wv_wsw  = P[10];
  const float* lne_b = P[67]; const float* lne_w = P[68];
  const float* lno_b = P[69]; const float* lno_w = P[70];
  const float* lnv_b = P[71]; const float* lnv_w = P[72];

  const size_t PS_HV  = (size_t)NN * 32;
  const size_t PS_EV  = (size_t)E;
  const size_t PS_AV  = (size_t)E * 96;
  const size_t PS_CVH = (size_t)E * 80;
  const size_t PS_CV  = (size_t)E * 32;
  const size_t PS_MV  = (size_t)E * 32;

  float*  hs    = allocF((size_t)NN * 128);
  float*  hv    = allocF(3 * PS_HV);
  bf16_t* hs_bf = allocB((size_t)NN * 128);
  bf16_t* hv_bf = allocB(3 * PS_HV);
  bf16_t* es_bf = allocB((size_t)E * 32);
  bf16_t* ev_bf = allocB(3 * PS_EV);
  float*  cnt   = allocF((size_t)NN);
  float*  acc_s = allocF((size_t)NN * 128);
  float*  acc_v = allocF(3 * PS_HV);
  float*  osum  = allocF(64);
  float*  ocnt  = allocF(16);
  bf16_t* A_s   = allocB((size_t)E * 384);
  bf16_t* A_v   = allocB(3 * PS_AV);     // also reused as vh_bf planes
  float*  C_vh  = allocF(3 * PS_CVH);
  float*  C_s   = allocF((size_t)E * 128);
  bf16_t* S_bf  = allocB((size_t)E * 128);
  float*  C_g   = allocF((size_t)E * 32);
  float*  C_v   = allocF(3 * PS_CV);
  bf16_t* mv_bf = allocB(3 * PS_MV);

  auto gemm = [&](const bf16_t* A, int lda, const bf16_t* B, int ldb,
                  float* C, int ldc, int M, int Kp, int Np) {
    int Mt = M / 16, Nt = Np / 16, Ks = Kp / 32;
    long long waves = (long long)Mt * Nt;
    gemm_bf16_wave<<<(unsigned)((waves + 7) / 8), 256, 0, stream>>>(
        A, lda, B, ldb, C, ldc, Mt, Nt, Ks);
  };

  // ---- init ----
  hipMemsetAsync(hv,    0, 3 * PS_HV * 4, stream);
  hipMemsetAsync(hv_bf, 0, 3 * PS_HV * 2, stream);
  hipMemsetAsync(cnt,   0, (size_t)NN * 4, stream);
  hipMemsetAsync(osum,  0, 64 * 4, stream);
  hipMemsetAsync(ocnt,  0, 16 * 4, stream);
  k_node_init<<<nblk(NN), 256, 0, stream>>>(node_attrs, lnv_w, lnv_b, Wv_wsw, Wv_wsb,
                                            hs, hs_bf, NN);
  k_edge_init<<<nblk(E), 256, 0, stream>>>(positions, src, dstp, shifts, lne_w, lne_b,
                                           We_wh, We_wsw, We_wsb, We_wsvw, We_wsvb, We_wv,
                                           es_bf, ev_bf, PS_EV, E);
  k_count<<<nblk(E), 256, 0, stream>>>(dstp, cnt, E);

  // ---- conv layers ----
  for (int l = 0; l < 2; ++l) {
    const LW& w = L[l];
    k_gather_s<<<nblk((long long)E * 384), 256, 0, stream>>>(hs_bf, es_bf, src, dstp, A_s, E);
    k_gather_v<<<nblk((long long)E * 288), 256, 0, stream>>>(hv_bf, PS_HV, ev_bf, PS_EV,
                                                             src, dstp, A_v, PS_AV, E);
    // msg0
    for (int c = 0; c < 3; ++c)
      gemm(A_v + c * PS_AV, 96, w.m0_wh, 80, C_vh + c * PS_CVH, 80, E, 96, 80);
    k_vn<<<nblk((long long)E * 65), 256, 0, stream>>>(C_vh, PS_CVH, 80, 65, A_s, 384, 288, E);
    k_cast_planes<<<nblk((long long)E * 288), 256, 0, stream>>>(C_vh, PS_CVH, 80, 65,
                                                                A_v, PS_AV, 96, E);
    gemm(A_s, 384, w.m0_ws, 128, C_s, 128, E, 384, 128);
    k_bias_act<<<nblk((long long)E * 128), 256, 0, stream>>>(C_s, w.m0_wsb, E, 128, 1, S_bf);
    gemm(S_bf, 128, w.m0_wsv, 32, C_g, 32, E, 128, 32);
    for (int c = 0; c < 3; ++c)
      gemm(A_v + c * PS_AV, 96, w.m0_wv, 32, C_v + c * PS_CV, 32, E, 96, 32);
    k_gate_mul<<<nblk((long long)E * 32), 256, 0, stream>>>(C_v, PS_CV, C_g, w.m0_wsvb,
                                                            E, 32, mv_bf, PS_MV);
    // msg1
    for (int c = 0; c < 3; ++c)
      gemm(mv_bf + c * PS_MV, 32, w.m1_wh, 32, C_vh + c * PS_CVH, 32, E, 32, 32);
    k_copy_bf<<<nblk((long long)E * 128), 256, 0, stream>>>(S_bf, 128, A_s, 160, E);
    k_vn<<<nblk((long long)E * 32), 256, 0, stream>>>(C_vh, PS_CVH, 32, 32, A_s, 160, 128, E);
    gemm(A_s, 160, w.m1_ws, 128, C_s, 128, E, 160, 128);
    k_bias_act<<<nblk((long long)E * 128), 256, 0, stream>>>(C_s, w.m1_wsb, E, 128, 0, S_bf);
    gemm(S_bf, 128, w.m1_wsv, 32, C_g, 32, E, 128, 32);
    k_cast_planes<<<nblk((long long)E * 96), 256, 0, stream>>>(C_vh, PS_CVH, 32, 32,
                                                               A_v, PS_AV, 32, E);
    for (int c = 0; c < 3; ++c)
      gemm(A_v + c * PS_AV, 32, w.m1_wv, 32, C_v + c * PS_CV, 32, E, 32, 32);
    k_gate_mul<<<nblk((long long)E * 32), 256, 0, stream>>>(C_v, PS_CV, C_g, w.m1_wsvb,
                                                            E, 32, (bf16_t*)nullptr, 0);
    // segment mean + residual + LN0
    hipMemsetAsync(acc_s, 0, (size_t)NN * 128 * 4, stream);
    hipMemsetAsync(acc_v, 0, 3 * PS_HV * 4, stream);
    k_scatter_add<<<nblk((long long)E * 128), 256, 0, stream>>>(C_s, dstp, acc_s, 128, E);
    for (int c = 0; c < 3; ++c)
      k_scatter_add<<<nblk((long long)E * 32), 256, 0, stream>>>(C_v + c * PS_CV, dstp,
                                                                 acc_v + c * PS_HV, 32, E);
    k_node_ln<<<nblk(NN), 256, 0, stream>>>(hs, hv, PS_HV, acc_s, acc_v, PS_HV, 32, cnt,
                                            w.ln0_w, w.ln0_b, hs_bf, hv_bf, PS_HV, NN);
    // ff0
    for (int c = 0; c < 3; ++c)
      gemm(hv_bf + c * PS_HV, 32, w.f0_wh, 64, C_vh + c * PS_CVH, 64, NN, 32, 64);
    k_copy_bf<<<nblk((long long)NN * 128), 256, 0, stream>>>(hs_bf, 128, A_s, 192, NN);
    k_vn<<<nblk((long long)NN * 64), 256, 0, stream>>>(C_vh, PS_CVH, 64, 64, A_s, 192, 128, NN);
    k_cast_planes<<<nblk((long long)NN * 192), 256, 0, stream>>>(C_vh, PS_CVH, 64, 64,
                                                                 A_v, PS_AV, 64, NN);
    gemm(A_s, 192, w.f0_ws, 512, C_s, 512, NN, 192, 512);
    k_bias_act<<<nblk((long long)NN * 512), 256, 0, stream>>>(C_s, w.f0_wsb, NN, 512, 1, S_bf);
    gemm(S_bf, 512, w.f0_wsv, 64, C_g, 64, NN, 512, 64);
    for (int c = 0; c < 3; ++c)
      gemm(A_v + c * PS_AV, 64, w.f0_wv, 64, C_v + c * PS_CV, 64, NN, 64, 64);
    k_gate_mul<<<nblk((long long)NN * 64), 256, 0, stream>>>(C_v, PS_CV, C_g, w.f0_wsvb,
                                                             NN, 64, mv_bf, PS_MV);
    // ff1
    for (int c = 0; c < 3; ++c)
      gemm(mv_bf + c * PS_MV, 64, w.f1_wh, 64, C_vh + c * PS_CVH, 64, NN, 64, 64);
    k_copy_bf<<<nblk((long long)NN * 512), 256, 0, stream>>>(S_bf, 512, A_s, 576, NN);
    k_vn<<<nblk((long long)NN * 64), 256, 0, stream>>>(C_vh, PS_CVH, 64, 64, A_s, 576, 512, NN);
    k_cast_planes<<<nblk((long long)NN * 192), 256, 0, stream>>>(C_vh, PS_CVH, 64, 64,
                                                                 A_v, PS_AV, 64, NN);
    gemm(A_s, 576, w.f1_ws, 128, C_s, 128, NN, 576, 128);
    k_bias_act<<<nblk((long long)NN * 128), 256, 0, stream>>>(C_s, w.f1_wsb, NN, 128, 0, S_bf);
    gemm(S_bf, 128, w.f1_wsv, 32, C_g, 32, NN, 128, 32);
    for (int c = 0; c < 3; ++c)
      gemm(A_v + c * PS_AV, 64, w.f1_wv, 32, C_v + c * PS_CV, 32, NN, 64, 32);
    k_gate_mul<<<nblk((long long)NN * 32), 256, 0, stream>>>(C_v, PS_CV, C_g, w.f1_wsvb,
                                                             NN, 32, (bf16_t*)nullptr, 0);
    k_node_ln<<<nblk(NN), 256, 0, stream>>>(hs, hv, PS_HV, C_s, C_v, PS_CV, 32,
                                            (const float*)nullptr,
                                            w.ln1_w, w.ln1_b, hs_bf, hv_bf, PS_HV, NN);
  }

  // ---- output head ----
  k_node_ln<<<nblk(NN), 256, 0, stream>>>(hs, hv, PS_HV,
                                          (const float*)nullptr, (const float*)nullptr, 0, 0,
                                          (const float*)nullptr, lno_w, lno_b,
                                          (bf16_t*)nullptr, (bf16_t*)nullptr, 0, NN);
  k_out_node<<<nblk(NN), 256, 0, stream>>>(hs, hv, PS_HV, Wo_wh, Wo_wsw, Wo_wsb,
                                           batch, osum, ocnt, NN);
  k_out_div<<<1, 64, 0, stream>>>(osum, ocnt, (float*)d_out);
}
